// IcoELLGATUNet_28793460752871
// MI455X (gfx1250) — compile-verified, matrix operand order checked by reading, not run on hardware
//
#include <hip/hip_runtime.h>
#include <hip/hip_bf16.h>
#include <stdint.h>

// ---------------------------------------------------------------------------
// IcoELLGAT U-Net forward for MI455X (gfx1250, wave32, WMMA + TDM).
//
// Per ellgat layer:
//   1. W (f32) -> Wh (f16);  Q (f32, KxN) -> QT (f16, NxK) transposed.
//   2. GEMM: WQ = W @ Q via v_wmma_f32_16x16x32_f16. A tile (16xKC of Wh) and
//      B tile (64xKC of QT) are DMA'd into LDS by the Tensor Data Mover
//      (tensor_load_to_lds + s_wait_tensorcnt), fragments built with
//      ds_load_b128 (k-contiguous rows in both tiles).
//   3. Attention (gather 7-8 neighbors, leaky-relu logits, softmax, weighted
//      aggregation) in fp32 VALU, one thread per (head, node).
// Everything is L2-resident (192MB L2 >> 32MB max feature map).
// ---------------------------------------------------------------------------

typedef __attribute__((ext_vector_type(16))) _Float16 v16h;
typedef __attribute__((ext_vector_type(8)))  _Float16 v8h;
typedef __attribute__((ext_vector_type(8)))  float    v8f;
typedef __attribute__((ext_vector_type(4)))  unsigned int u32x4;
typedef __attribute__((ext_vector_type(8)))  int      i32x8;
typedef __attribute__((ext_vector_type(4)))  int      i32x4;

#define GEMM_THREADS 128   // 4 waves of 32
#define KC 128             // K-chunk staged in LDS (four WMMA k-steps of 32)

#if __has_builtin(__builtin_amdgcn_tensor_load_to_lds) && \
    __has_builtin(__builtin_amdgcn_s_wait_tensorcnt)
#define HAVE_TDM 1
#else
#define HAVE_TDM 0
#endif

#if HAVE_TDM
// Issue one 2D TDM tile load: tile_d1 rows x tile_d0 cols of f16 elements,
// from row-major tensor with row stride `stride0` (elements), into LDS at
// byte offset lds_off (compact row-major tile). `tensor_d0/d1` are the
// REMAINING extents measured from the tile origin, so the TDM's OOB rule
// (reads beyond tensor dim return zero) implements our K/N-edge zero padding.
// D# packing per CDNA5 ISA (group0) / §8.4 (group1).
// This toolchain exposes the 6-arg builtin (extra int32x8 group before cpol).
__device__ __forceinline__ void tdm_load_2d_f16(
    unsigned lds_off, const _Float16* gaddr,
    unsigned tensor_d0, unsigned tensor_d1, unsigned stride0,
    unsigned tile_d0, unsigned tile_d1)
{
    unsigned long long ga = (unsigned long long)(uintptr_t)gaddr;
    u32x4 g0;
    g0[0] = 1u;                                       // count=1, user descriptor
    g0[1] = lds_off;                                  // lds_addr [63:32]
    g0[2] = (unsigned)(ga & 0xFFFFFFFFu);             // global_addr[31:0]
    g0[3] = (unsigned)((ga >> 32) & 0x01FFFFFFu)      // global_addr[56:32]
          | (2u << 30);                               // type=2 ("image")
    i32x8 g1;
    g1[0] = (int)(1u << 16);                          // wg_mask=0, data_size=1 (2B)
    g1[1] = (int)((tensor_d0 & 0xFFFFu) << 16);       // tensor_dim0[15:0] @ bits63:48
    g1[2] = (int)((tensor_d0 >> 16) |                 // tensor_dim0[31:16]
                  ((tensor_d1 & 0xFFFFu) << 16));     // tensor_dim1[15:0]
    g1[3] = (int)((tensor_d1 >> 16) |                 // tensor_dim1[31:16]
                  (tile_d0 << 16));                   // tile_dim0
    g1[4] = (int)(tile_d1 & 0xFFFFu);                 // tile_dim1 (tile_dim2=0)
    g1[5] = (int)stride0;                             // tensor_dim0_stride[31:0]
    g1[6] = 0;                                        // stride0 hi | stride1 lo
    g1[7] = 0;
    i32x4 z4 = {0, 0, 0, 0};
    i32x8 z8 = {0, 0, 0, 0, 0, 0, 0, 0};
    __builtin_amdgcn_tensor_load_to_lds(g0, g1, z4, z4, z8, 0);
}
#endif

// out[M x N](f32) = Wh[M x K](f16) @ QT[N x K](f16, transposed activations).
// Block = 16(M) x 64(N) tile, 4 waves, one 16x16 WMMA accumulator per wave.
__global__ __launch_bounds__(GEMM_THREADS)
void gemm_wmma_tdm_kernel(const _Float16* __restrict__ Wh,
                          const _Float16* __restrict__ QT,
                          float* __restrict__ out, int M, int Kd, int Nn)
{
    __shared__ _Float16 lA[16][KC];    // A tile: rows k-contiguous
    __shared__ _Float16 lBt[64][KC];   // B tile (transposed): rows k-contiguous

    const int tid  = threadIdx.x;
    const int wave = tid >> 5;
    const int lane = tid & 31;
    const int half = lane >> 4;
    const int lr   = lane & 15;
    const int mBase = blockIdx.y * 16;
    const int nBase = blockIdx.x * 64;

    v8f acc = {};

#if HAVE_TDM
    const unsigned ldsA = (unsigned)(uintptr_t)&lA[0][0];
    const unsigned ldsB = (unsigned)(uintptr_t)&lBt[0][0];
#endif

    for (int k0 = 0; k0 < Kd; k0 += KC) {
#if HAVE_TDM
        if (tid < 32) {   // wave 0 drives the DMA; TDM ignores EXEC per-lane
            tdm_load_2d_f16(ldsA, Wh + (size_t)mBase * Kd + k0,
                            (unsigned)(Kd - k0), (unsigned)(M - mBase),
                            (unsigned)Kd, KC, 16);
            tdm_load_2d_f16(ldsB, QT + (size_t)nBase * Kd + k0,
                            (unsigned)(Kd - k0),
                            (unsigned)(Nn - nBase > 64 ? 64 : Nn - nBase),
                            (unsigned)Kd, KC, 64);
            __builtin_amdgcn_s_wait_tensorcnt(0);
        }
        __syncthreads();
#else
        for (int idx = tid; idx < 16 * KC; idx += GEMM_THREADS) {
            int r = idx / KC, c = idx % KC;
            int kk = k0 + c;
            lA[r][c] = (kk < Kd) ? Wh[(size_t)(mBase + r) * Kd + kk] : (_Float16)0.0f;
        }
        for (int idx = tid; idx < 64 * KC; idx += GEMM_THREADS) {
            int r = idx / KC, c = idx % KC;
            int kk = k0 + c, nn = nBase + r;
            lBt[r][c] = (kk < Kd && nn < Nn) ? QT[(size_t)nn * Kd + kk] : (_Float16)0.0f;
        }
        __syncthreads();
#endif

        #pragma unroll
        for (int kk = 0; kk < KC; kk += 32) {
            // 16-bit 16x32 A layout: lanes 0-15 (half=0) hold K 0..7 / 16..23,
            // lanes 16-31 (half=1) hold K 8..15 / 24..31 -> two contiguous
            // 16-byte chunks per fragment -> ds_load_b128.
            const v8h a0 = *(const v8h*)&lA[lr][kk + half * 8];
            const v8h a1 = *(const v8h*)&lA[lr][kk + 16 + half * 8];
            const v8h b0 = *(const v8h*)&lBt[wave * 16 + lr][kk + half * 8];
            const v8h b1 = *(const v8h*)&lBt[wave * 16 + lr][kk + 16 + half * 8];
            v16h af = __builtin_shufflevector(a0, a1, 0,1,2,3,4,5,6,7,8,9,10,11,12,13,14,15);
            v16h bf = __builtin_shufflevector(b0, b1, 0,1,2,3,4,5,6,7,8,9,10,11,12,13,14,15);
            acc = __builtin_amdgcn_wmma_f32_16x16x32_f16(
                false, af, false, bf, (short)0, acc, false, false);
        }
        __syncthreads();
    }

    // C/D layout: VGPR i -> row (i + 8*half), col = lane&15 within wave tile.
    const int col = nBase + wave * 16 + lr;
    if (col < Nn) {
        #pragma unroll
        for (int i = 0; i < 8; ++i) {
            int m = mBase + i + 8 * half;
            if (m < M) out[(size_t)m * Nn + col] = acc[i];
        }
    }
}

// Wh[i] = (f16)W[i]
__global__ void cvt_f16_kernel(const float* __restrict__ s, _Float16* __restrict__ d, long n)
{
    long i = (long)blockIdx.x * blockDim.x + threadIdx.x;
    if (i < n) d[i] = (_Float16)s[i];
}

// QT[n][k] = (f16)Q[k][n]   (K x N row-major -> N x K row-major)
__global__ void transpose_cvt_f16_kernel(const float* __restrict__ src,
                                         _Float16* __restrict__ dst, int Kd, int Nn)
{
    long i = (long)blockIdx.x * blockDim.x + threadIdx.x;
    if (i >= (long)Kd * Nn) return;
    int k = (int)(i / Nn), n = (int)(i % Nn);
    dst[(long)n * Kd + k] = (_Float16)src[i];
}

// ek[h][n] = sum_f a_dst[h][f] * wk[h*f+f][n]
__global__ void ek_kernel(const float* __restrict__ a, const float* __restrict__ wk,
                          float* __restrict__ ek, int h, int f, int nK)
{
    int i = blockIdx.x * blockDim.x + threadIdx.x;
    if (i >= h * nK) return;
    int hh = i / nK, n = i % nK;
    const float* ad = a + (long)hh * 2 * f;
    float s = 0.f;
    for (int ff = 0; ff < f; ++ff)
        s += ad[ff] * wk[(long)(hh * f + ff) * nK + n];
    ek[i] = s;
}

__device__ __forceinline__ float lrelu(float x) { return x > 0.f ? x : 0.01f * x; }

// Per (head, node): gather neighbors, logits, softmax, weighted aggregation.
__global__ void gat_attn_kernel(const float* __restrict__ a, const float* __restrict__ wq,
                                const float* __restrict__ ek, const int* __restrict__ adj,
                                float* __restrict__ out,
                                int h, int f, int nQ, int nK, int kadj, int relu_out)
{
    int i = blockIdx.x * blockDim.x + threadIdx.x;
    if (i >= h * nK) return;
    int hh = i / nK, n = i % nK;
    const float* as = a + (long)hh * 2 * f + f;
    const int* arow = adj + (long)n * kadj;
    __builtin_prefetch(arow, 0, 0);   // global_prefetch_b8

    int   jj[8];
    float lg[8];
    const float ekv = ek[(long)hh * nK + n];
    float mx = -__builtin_inff();
    for (int k = 0; k < kadj; ++k) {
        int j = arow[k];
        jj[k] = j;
        float e;
        if (j >= 0) {
            const float* col = wq + (long)(hh * f) * nQ + j;
            float s = 0.f;
            for (int ff = 0; ff < f; ++ff) s += as[ff] * col[(long)ff * nQ];
            e = lrelu(ekv + s);
        } else {
            e = -__builtin_inff();
        }
        lg[k] = e;
        mx = fmaxf(mx, e);
    }
    float ssum = 0.f;
    for (int k = 0; k < kadj; ++k) {
        float v = __expf(lg[k] - mx);
        lg[k] = v;
        ssum += v;
    }
    const float inv = 1.0f / ssum;
    for (int ff = 0; ff < f; ++ff) {
        const float* rowp = wq + (long)(hh * f + ff) * nQ;
        float accv = 0.f;
        for (int k = 0; k < kadj; ++k)
            if (jj[k] >= 0) accv += lg[k] * rowp[jj[k]];
        accv *= inv;
        out[(long)(hh * f + ff) * nK + n] = relu_out ? lrelu(accv) : accv;
    }
}

__global__ void zero_kernel(float* __restrict__ p, long n)
{
    long i = (long)blockIdx.x * blockDim.x + threadIdx.x;
    if (i < n) p[i] = 0.f;
}

__global__ void scatter_add_feat(const float* __restrict__ Q, const int* __restrict__ argadj,
                                 float* __restrict__ Qm, int c, int nQ, int nK)
{
    long i = (long)blockIdx.x * blockDim.x + threadIdx.x;
    if (i >= (long)c * nQ) return;
    int ch = (int)(i / nQ), j = (int)(i % nQ);
    atomicAdd(&Qm[(long)ch * nK + argadj[j]], Q[i]);
}

__global__ void scatter_count(const int* __restrict__ argadj, float* __restrict__ cnt, int nQ)
{
    int i = blockIdx.x * blockDim.x + threadIdx.x;
    if (i < nQ) atomicAdd(&cnt[argadj[i]], 1.0f);
}

__global__ void divide_kernel(float* __restrict__ Qm, const float* __restrict__ cnt,
                              int c, int nK)
{
    long i = (long)blockIdx.x * blockDim.x + threadIdx.x;
    if (i >= (long)c * nK) return;
    float ct = cnt[i % nK];
    Qm[i] = (ct == 0.f) ? 0.f : Qm[i] / ct;
}

// out[(cA+cB) x n] = concat(A[cA x n], B[cB x nSrc] gathered via argadj[n])
__global__ void concat_gather_kernel(const float* __restrict__ A, const float* __restrict__ B,
                                     const int* __restrict__ argadj, float* __restrict__ out,
                                     int cA, int cB, int n, int nSrc)
{
    long i = (long)blockIdx.x * blockDim.x + threadIdx.x;
    if (i >= (long)(cA + cB) * n) return;
    int row = (int)(i / n), col = (int)(i % n);
    out[i] = (row < cA) ? A[(long)row * n + col]
                        : B[(long)(row - cA) * nSrc + argadj[col]];
}

// ---------------------------------------------------------------------------
// Host-side orchestration
// ---------------------------------------------------------------------------
static inline int ceil_div(int a, int b) { return (a + b - 1) / b; }

struct Bufs { float *T, *WQ, *WK, *EK, *CNT; _Float16 *Wh, *QT; };

static void run_gemm(const float* W, const float* Qsrc, float* out,
                     int M, int Kd, int Nn, Bufs& bufs, hipStream_t s)
{
    long wn = (long)M * Kd;
    cvt_f16_kernel<<<ceil_div((int)wn, 256), 256, 0, s>>>(W, bufs.Wh, wn);
    long qn = (long)Kd * Nn;
    transpose_cvt_f16_kernel<<<ceil_div((int)qn, 256), 256, 0, s>>>(Qsrc, bufs.QT, Kd, Nn);
    dim3 g(ceil_div(Nn, 64), M / 16), b(GEMM_THREADS);
    gemm_wmma_tdm_kernel<<<g, b, 0, s>>>(bufs.Wh, bufs.QT, out, M, Kd, Nn);
}

static void run_ellgat(const float* W, const float* a, int h, int f, int cin,
                       const float* Qf, int nQ, const float* Kf, int nK,
                       const int* adj, int kadj, float* out, int relu,
                       Bufs& bufs, hipStream_t s)
{
    const int M = h * f;
    run_gemm(W, Qf, bufs.WQ, M, cin, nQ, bufs, s);
    const float* wk = bufs.WQ;
    if (Kf != Qf) { run_gemm(W, Kf, bufs.WK, M, cin, nK, bufs, s); wk = bufs.WK; }
    int tot = h * nK;
    ek_kernel<<<ceil_div(tot, 256), 256, 0, s>>>(a, wk, bufs.EK, h, f, nK);
    gat_attn_kernel<<<ceil_div(tot, 128), 128, 0, s>>>(a, bufs.WQ, bufs.EK, adj,
                                                       out, h, f, nQ, nK, kadj, relu);
}

static void run_block(const float* W1, const float* a1, int h1, int f1, int cin1,
                      const float* W2, const float* a2, int h2, int f2,
                      const float* Qin, int n, const int* adj, int kadj,
                      float* out, Bufs& bufs, hipStream_t s)
{
    run_ellgat(W1, a1, h1, f1, cin1, Qin, n, Qin, n, adj, kadj, bufs.T, 1, bufs, s);
    run_ellgat(W2, a2, h2, f2, h1 * f1, bufs.T, n, bufs.T, n, adj, kadj, out, 1, bufs, s);
}

static void run_resample(const float* W, const float* a, int h, int f, int c,
                         const float* Q, int nQ, const int* argadj, const int* bip,
                         int nK, float* out, float* Qm, Bufs& bufs, hipStream_t s)
{
    long e1 = (long)c * nK;
    zero_kernel<<<ceil_div((int)e1, 256), 256, 0, s>>>(Qm, e1);
    zero_kernel<<<ceil_div(nK, 256), 256, 0, s>>>(bufs.CNT, (long)nK);
    long e2 = (long)c * nQ;
    scatter_add_feat<<<ceil_div((int)e2, 256), 256, 0, s>>>(Q, argadj, Qm, c, nQ, nK);
    scatter_count<<<ceil_div(nQ, 256), 256, 0, s>>>(argadj, bufs.CNT, nQ);
    divide_kernel<<<ceil_div((int)e1, 256), 256, 0, s>>>(Qm, bufs.CNT, c, nK);
    run_ellgat(W, a, h, f, c, Q, nQ, Qm, nK, bip, 8, out, 1, bufs, s);
}

extern "C" void kernel_launch(void* const* d_in, const int* in_sizes, int n_in,
                              void* d_out, int out_size, void* d_ws, size_t ws_size,
                              hipStream_t stream)
{
    // params: 15 layers x (W, a) in dict order:
    // c0_1 c0_2 c1_1 c1_2 c2_1 c2_2 r0 r1 e0_1 e0_2 e1_1 e1_2 e2_1 e2_2 readout
    const float* W[15]; const float* A[15];
    for (int i = 0; i < 15; ++i) {
        W[i] = (const float*)d_in[2 * i];
        A[i] = (const float*)d_in[2 * i + 1];
    }
    const float* X0    = (const float*)d_in[30];
    const int*   adj0  = (const int*)d_in[31];
    const int*   adj1  = (const int*)d_in[32];
    const int*   adj2  = (const int*)d_in[33];
    const int*   bip01 = (const int*)d_in[34];
    const int*   bip12 = (const int*)d_in[35];
    const int*   arg01 = (const int*)d_in[36];
    const int*   arg12 = (const int*)d_in[37];
    float* out = (float*)d_out;

    const int N0 = 40962, N1 = 10242, N2 = 2562;

    // Workspace carve
    float* ws = (float*)d_ws;
    size_t off = 0;
    auto carve = [&](size_t n) { float* p = ws + off; off += n; return p; };
    float* P0    = carve((size_t)192 * N0 + 64);
    float* P1    = carve((size_t)192 * N0 + 64);
    Bufs bufs;
    bufs.T   = carve((size_t)64 * N0 + 64);
    bufs.WQ  = carve((size_t)64 * N0 + 64);
    bufs.WK  = carve((size_t)64 * N2 + 64 * 16384);
    bufs.EK  = carve((size_t)4 * N0 + 64);
    bufs.CNT = carve((size_t)N0 + 64);
    float* skip0 = carve((size_t)64 * N0 + 64);
    float* skip1 = carve((size_t)128 * N1 + 64);
    float* skip2 = carve((size_t)256 * N2 + 64);
    float* Qm    = carve((size_t)128 * N1 + 64);
    // f16 staging region (Wh: <=256x384, QT: <=192 x N0 transposed)
    _Float16* h16 = (_Float16*)(ws + off);
    bufs.Wh = h16;
    bufs.QT = h16 + (256 * 384 + 64);

    // ---- Encoder ----
    run_block(W[0], A[0], 4, 16, 16,  W[1], A[1], 4, 16, X0,    N0, adj0, 7, skip0, bufs, stream); // 64 x N0
    run_resample(W[6], A[6], 4, 8, 64,  skip0, N0, arg01, bip01, N1, P0, Qm, bufs, stream);        // 32 x N1
    run_block(W[2], A[2], 4, 32, 32,  W[3], A[3], 4, 32, P0,    N1, adj1, 7, skip1, bufs, stream); // 128 x N1
    run_resample(W[7], A[7], 4, 16, 128, skip1, N1, arg12, bip12, N2, P0, Qm, bufs, stream);       // 64 x N2
    run_block(W[4], A[4], 4, 64, 64,  W[5], A[5], 4, 64, P0,    N2, adj2, 7, skip2, bufs, stream); // 256 x N2

    // ---- Bottleneck expand (e2) ----
    run_block(W[12], A[12], 4, 64, 256, W[13], A[13], 4, 64, skip2, N2, adj2, 7, P0, bufs, stream); // 256 x N2

    // ---- Decoder level 1: concat [skip1(128); up(256)] -> 384 x N1 ----
    {
        long tot = (long)(128 + 256) * N1;
        concat_gather_kernel<<<ceil_div((int)tot, 256), 256, 0, stream>>>(
            skip1, P0, arg12, P1, 128, 256, N1, N2);
    }
    run_block(W[10], A[10], 4, 32, 384, W[11], A[11], 4, 32, P1, N1, adj1, 7, P0, bufs, stream);   // 128 x N1

    // ---- Decoder level 0: concat [skip0(64); up(128)] -> 192 x N0 ----
    {
        long tot = (long)(64 + 128) * N0;
        concat_gather_kernel<<<ceil_div((int)tot, 256), 256, 0, stream>>>(
            skip0, P0, arg01, P1, 64, 128, N0, N1);
    }
    run_block(W[8], A[8], 4, 16, 192, W[9], A[9], 4, 16, P1, N0, adj0, 7, P0, bufs, stream);       // 64 x N0

    // ---- Readout (h=1, f=32, no trailing nonlinearity) ----
    run_ellgat(W[14], A[14], 1, 32, 64, P0, N0, P0, N0, adj0, 7, out, 0, bufs, stream);            // 32 x N0
}
